// PaiNNMixing_88897233093049
// MI455X (gfx1250) — matrix operand order repaired and checked by reading, not compiled, not run on hardware
//
#include <hip/hip_runtime.h>
#include <hip/hip_bf16.h>
#include <math.h>

// ---------------------------------------------------------------------------
// Fused PaiNN mixing block for gfx1250 (MI455X), wave32 + WMMA bf16.
//
//   Pre-pass (once per launch, if ws_size allows): convert U_W/V_W/W1/W2 to
//   bf16, transposed to [n][k] (K contiguous) in d_ws -> halves L2 weight
//   traffic, removes per-block f32->bf16 conversion, enables async LDS DMA.
//
//   Main kernel (1 block = 16 atoms, 256 threads = 8 wave32):
//     Uv = v @ U_W ; Vv = v @ V_W            (bf16 WMMA, f32 accum, K=256)
//     Vv_norm = sqrt(sum_d Vv^2 + 1e-8)      (LDS f32 atomics over d)
//     ctx_in  = [LayerNorm(s) | Vv_norm]     (built in LDS as bf16)
//     X = silu(ctx_in @ W1 + b1)             (WMMA, K=512)
//     Y = X @ W2 + b2                        (WMMA, K=256, 768 cols)
//     s_out = s + a_ss + a_sv * (Uv.Vv)
//     v_out = v + a_vv * Uv                  (Uv kept in registers)
// ---------------------------------------------------------------------------

typedef __bf16 bf16;
typedef __attribute__((ext_vector_type(16))) __bf16 v16bf;
typedef __attribute__((ext_vector_type(8)))  __bf16 v8bf;
typedef __attribute__((ext_vector_type(8)))  float  v8f;
typedef __attribute__((ext_vector_type(4)))  int    v4i_;

// CDNA5 async global->LDS DMA (ASYNCcnt), guarded so the file always compiles.
#if defined(__HIP_DEVICE_COMPILE__) && defined(__has_builtin)
#if __has_builtin(__builtin_amdgcn_global_load_async_to_lds_b128) && \
    __has_builtin(__builtin_amdgcn_s_wait_asynccnt)
#define HAVE_ASYNC_LDS 1
#endif
#endif

#if HAVE_ASYNC_LDS
typedef __attribute__((address_space(1))) v4i_ ga_v4i;   // global int4
typedef __attribute__((address_space(3))) v4i_ ls_v4i;   // LDS int4
#endif

__device__ inline void async_copy16(const bf16* g, bf16* l) {
#if HAVE_ASYNC_LDS
  __builtin_amdgcn_global_load_async_to_lds_b128((ga_v4i*)g, (ls_v4i*)l, 0, 0);
#else
  *(v8bf*)l = *(const v8bf*)g;
#endif
}
__device__ inline void async_join() {
#if HAVE_ASYNC_LDS
  __builtin_amdgcn_s_wait_asynccnt(0);
#endif
}

#define H   256
#define MT  16           // atoms per block
#define VROWS 48         // 3*MT

#define SA_STRIDE  264   // bf16 elems per row (pad -> bank spread, 16B aligned)
#define CTX_STRIDE 520
#define XA_STRIDE  264

// 64KB static LDS, regions time-shared across phases (byte offsets):
#define OFF_SA   0       // 48*264*2  = 25344   (phase 1 A: v tile, bf16)
#define OFF_WT   32768   // 256*32*2  = 16384   (phase 1 B staging)
#define OFF_DOT  0       // 16*256*4  = 16384   (Uv.Vv, persists to the end)
#define OFF_NRM  49152   // 16*256*4  = 16384   (sum Vv^2, dead after ctx build)
#define OFF_CTX  25344   // 16*520*2  = 16640   (ctx_in bf16, phase 2 A)
#define OFF_W1T  41984   // 256*32*2  = 16384   (phase 2 B staging)
#define OFF_XA   49152   // 16*264*2  =  8448   (silu(X) bf16, phase 3 A)
#define OFF_W2T  25344   // 256*32*2  = 16384   (phase 3 B staging)
#define OFF_AVV  49152   // 16*256*4  = 16384   (a_vv f32, for v_out phase)

// Workspace layout for pre-converted bf16 transposed weights (elements):
#define WSE_U   0               // [256][256]
#define WSE_V   65536           // [256][256]
#define WSE_W1  131072          // [256][512]
#define WSE_W2  262144          // [768][256]
#define WS_NEED_BYTES ((262144 + 196608) * 2)   // 917504

__device__ inline v8f wmma_bf16(v16bf a, v16bf b, v8f c) {
  return __builtin_amdgcn_wmma_f32_16x16x32_bf16(false, a, false, b,
                                                 (short)0, c, false, false);
}

// A fragment (16x32 bf16): lanes 0-15 hold K{k0..k0+7, k0+16..k0+23},
// lanes 16-31 hold K{k0+8..k0+15, k0+24..k0+31}  -> two aligned 16B LDS loads.
__device__ inline v16bf load_fragA(const bf16* base, int stride,
                                   int rowBase, int kBase, int lane) {
  const bf16* q = base + (rowBase + (lane & 15)) * stride + kBase
                       + ((lane >> 4) << 3);
  v8bf lo = *(const v8bf*)q;
  v8bf hi = *(const v8bf*)(q + 16);
  v16bf r;
#pragma unroll
  for (int i = 0; i < 8; ++i) { r[i] = lo[i]; r[i + 8] = hi[i]; }
  return r;
}

// B fragment (32x16 bf16) from LDS staged as Wt[n][k]: lane n = nBase+lane%16,
// lanes<16 -> K{0..7,16..23}, lanes>=16 -> K{8..15,24..31} (two aligned 16B loads).
__device__ inline v16bf load_fragB(const bf16* wt, int nBase, int lane) {
  const bf16* q = wt + (nBase + (lane & 15)) * 32 + ((lane >> 4) << 3);
  v8bf lo = *(const v8bf*)q;
  v8bf hi = *(const v8bf*)(q + 16);
  v16bf r;
#pragma unroll
  for (int i = 0; i < 8; ++i) { r[i] = lo[i]; r[i + 8] = hi[i]; }
  return r;
}

// Stage one 32-K x 256-N weight slab into LDS as sWt[n*32 + krel].
// PRE: pure copy from pre-transposed bf16 ws (async DMA if available).
// !PRE: gather from f32 row-major weights + convert.
template <bool PRE>
__device__ inline void stage_block(bf16* dst, const float* __restrict__ Wf,
                                   const bf16* __restrict__ Wt, int ldw, int Kt,
                                   int colBase, int kRow0, int tid) {
  if (PRE) {
    const bf16* src = Wt + (long)(colBase + tid) * Kt + kRow0;  // 64B aligned
    bf16* d = dst + tid * 32;
#pragma unroll
    for (int c = 0; c < 4; ++c) async_copy16(src + 8 * c, d + 8 * c);
  } else {
#pragma unroll 8
    for (int krel = 0; krel < 32; ++krel)
      dst[tid * 32 + krel] = (bf16)Wf[(long)(kRow0 + krel) * ldw + colBase + tid];
  }
}

template <bool PRE>
__global__ __launch_bounds__(256)
void painn_fused_kernel(const float* __restrict__ s, const float* __restrict__ v,
                        const float* __restrict__ gamma, const float* __restrict__ beta,
                        const float* __restrict__ U_W, const float* __restrict__ V_W,
                        const float* __restrict__ W1, const float* __restrict__ b1,
                        const float* __restrict__ W2, const float* __restrict__ b2,
                        const bf16* __restrict__ wsU, const bf16* __restrict__ wsV,
                        const bf16* __restrict__ wsW1, const bf16* __restrict__ wsW2,
                        float* __restrict__ s_out, float* __restrict__ v_out,
                        int nAtoms) {
  __shared__ __align__(16) unsigned char smem[65536];
  bf16*  sA   = (bf16*)(smem + OFF_SA);
  bf16*  sWt  = (bf16*)(smem + OFF_WT);
  float* dotS = (float*)(smem + OFF_DOT);
  float* nrmS = (float*)(smem + OFF_NRM);
  bf16*  ctxA = (bf16*)(smem + OFF_CTX);
  bf16*  sW1t = (bf16*)(smem + OFF_W1T);
  bf16*  xA   = (bf16*)(smem + OFF_XA);
  bf16*  sW2t = (bf16*)(smem + OFF_W2T);
  float* avvS = (float*)(smem + OFF_AVV);

  const int tid   = threadIdx.x;
  const int lane  = tid & 31;
  const int wave  = tid >> 5;
  const int atom0 = blockIdx.x * MT;
  const long vBase = (long)atom0 * 3 * H;

  // ---- Phase 0: v tile -> LDS bf16 (row = 3*a + d, linear in memory) ------
  for (int row = 0; row < VROWS; ++row) {
    float f = (atom0 + row / 3 < nAtoms) ? v[vBase + (long)row * H + tid] : 0.f;
    sA[row * SA_STRIDE + tid] = (bf16)f;
  }
  __syncthreads();

  // ---- Phase 1: Uv, Vv = v @ {U_W,V_W}; 48x256 tiles, K=256 ---------------
  // wave w owns tiles t = w, w+8, ... (t<48): rt = t%3, ct = t/3.
  v8f accU[6], accV[6];
#pragma unroll
  for (int i = 0; i < 6; ++i)
#pragma unroll
    for (int j = 0; j < 8; ++j) { accU[i][j] = 0.f; accV[i][j] = 0.f; }

  for (int mat = 0; mat < 2; ++mat) {
    const float* Wf = mat ? V_W : U_W;
    const bf16*  Wt = mat ? wsV : wsU;
    for (int ks = 0; ks < 8; ++ks) {
      __syncthreads();
      if (!PRE) __builtin_prefetch(&Wf[(long)((ks + 1) & 7) * 32 * H], 0, 0);
      stage_block<PRE>(sWt, Wf, Wt, H, H, 0, ks * 32, tid);
      async_join();
      __syncthreads();
#pragma unroll
      for (int i = 0; i < 6; ++i) {
        int t = wave + 8 * i;
        int rt = t % 3, ct = t / 3;
        v16bf a = load_fragA(sA, SA_STRIDE, rt * 16, ks * 32, lane);
        v16bf b = load_fragB(sWt, ct * 16, lane);
        if (mat == 0) accU[i] = wmma_bf16(a, b, accU[i]);
        else          accV[i] = wmma_bf16(a, b, accV[i]);
      }
    }
  }
  __syncthreads();

  // ---- Reduction over d: dot_uv and sum(Vv^2) via LDS f32 atomics ---------
#pragma unroll
  for (int it = 0; it < 16; ++it) dotS[it * 256 + tid] = 0.f;
#pragma unroll
  for (int it = 0; it < 16; ++it) nrmS[it * 256 + tid] = 0.f;
  __syncthreads();
  for (int i = 0; i < 6; ++i) {
    int t = wave + 8 * i, rt = t % 3, ct = t / 3;
    int colb = ct * 16 + (lane & 15);
#pragma unroll
    for (int j = 0; j < 8; ++j) {
      int row  = rt * 16 + j + ((lane >> 4) << 3);   // < 48 always
      int atom = row / 3;
      float uv = accU[i][j], vv = accV[i][j];
      atomicAdd(&nrmS[atom * H + colb], vv * vv);
      atomicAdd(&dotS[atom * H + colb], uv * vv);
    }
  }
  __syncthreads();

  // ---- LayerNorm(s) + Vv_norm -> ctx_in (bf16, 16x512); 2 atoms / wave ----
#pragma unroll
  for (int rep = 0; rep < 2; ++rep) {
    int a  = wave * 2 + rep;
    int ga = atom0 + a;
    float x[8];
    float sum = 0.f;
#pragma unroll
    for (int q = 0; q < 8; ++q) {
      int col = lane + 32 * q;
      x[q] = (ga < nAtoms) ? s[(long)ga * H + col] : 0.f;
      sum += x[q];
    }
    for (int off = 16; off > 0; off >>= 1) sum += __shfl_xor(sum, off, 32);
    float mu = sum * (1.f / 256.f);
    float vs = 0.f;
#pragma unroll
    for (int q = 0; q < 8; ++q) { float d = x[q] - mu; vs += d * d; }
    for (int off = 16; off > 0; off >>= 1) vs += __shfl_xor(vs, off, 32);
    float rstd = rsqrtf(vs * (1.f / 256.f) + 1e-5f);
#pragma unroll
    for (int q = 0; q < 8; ++q) {
      int col = lane + 32 * q;
      float sn = (x[q] - mu) * rstd * gamma[col] + beta[col];
      ctxA[a * CTX_STRIDE + col]     = (bf16)sn;
      ctxA[a * CTX_STRIDE + H + col] = (bf16)sqrtf(nrmS[a * H + col] + 1e-8f);
    }
  }
  __syncthreads();

  // ---- Phase 2: X = ctx_in @ W1 (16x256, K=512); wave w -> cols 32w..+31 --
  v8f accX[2];
#pragma unroll
  for (int c = 0; c < 2; ++c)
#pragma unroll
    for (int j = 0; j < 8; ++j) accX[c][j] = 0.f;

  for (int ks = 0; ks < 16; ++ks) {
    __syncthreads();
    stage_block<PRE>(sW1t, W1, wsW1, H, 2 * H, 0, ks * 32, tid);
    async_join();
    __syncthreads();
    v16bf a = load_fragA(ctxA, CTX_STRIDE, 0, ks * 32, lane);
#pragma unroll
    for (int c = 0; c < 2; ++c) {
      v16bf b = load_fragB(sW1t, (wave * 2 + c) * 16, lane);
      accX[c] = wmma_bf16(a, b, accX[c]);
    }
  }
  __syncthreads();

  // silu(X + b1) -> xA (bf16)
#pragma unroll
  for (int c = 0; c < 2; ++c) {
    int colb = (wave * 2 + c) * 16 + (lane & 15);
    float bias = b1[colb];
#pragma unroll
    for (int j = 0; j < 8; ++j) {
      int row  = j + ((lane >> 4) << 3);
      float xv = accX[c][j] + bias;
      float sl = xv / (1.f + __expf(-xv));
      xA[row * XA_STRIDE + colb] = (bf16)sl;
    }
  }
  __syncthreads();

  // ---- Phase 3: Y = silu(X) @ W2 (16x768, K=256) --------------------------
  v8f accY[6];
#pragma unroll
  for (int c = 0; c < 6; ++c)
#pragma unroll
    for (int j = 0; j < 8; ++j) accY[c][j] = 0.f;

  for (int nb = 0; nb < 3; ++nb) {
    for (int ks = 0; ks < 8; ++ks) {
      __syncthreads();
      stage_block<PRE>(sW2t, W2, wsW2, 3 * H, H, nb * H, ks * 32, tid);
      async_join();
      __syncthreads();
      v16bf a = load_fragA(xA, XA_STRIDE, 0, ks * 32, lane);
#pragma unroll
      for (int c = 0; c < 2; ++c) {
        v16bf b = load_fragB(sW2t, (wave * 2 + c) * 16, lane);
        accY[nb * 2 + c] = wmma_bf16(a, b, accY[nb * 2 + c]);
      }
    }
  }
  __syncthreads();

  // ---- Epilogue: s_out, a_vv to LDS ---------------------------------------
#pragma unroll
  for (int c = 0; c < 2; ++c) {
    int colb = (wave * 2 + c) * 16 + (lane & 15);
    float bss = b2[colb], bsv = b2[H + colb], bvv = b2[2 * H + colb];
#pragma unroll
    for (int j = 0; j < 8; ++j) {
      int a = j + ((lane >> 4) << 3);
      avvS[a * H + colb] = accY[4 + c][j] + bvv;
      int ga = atom0 + a;
      if (ga < nAtoms) {
        float a_ss = accY[c][j]     + bss;
        float a_sv = accY[2 + c][j] + bsv;
        long  o    = (long)ga * H + colb;
        s_out[o] = s[o] + a_ss + a_sv * dotS[a * H + colb];
      }
    }
  }
  __syncthreads();

  // ---- v_out = v + a_vv * Uv (Uv still in phase-1 registers) --------------
  for (int i = 0; i < 6; ++i) {
    int t = wave + 8 * i, rt = t % 3, ct = t / 3;
    int colb = ct * 16 + (lane & 15);
#pragma unroll
    for (int j = 0; j < 8; ++j) {
      int row  = rt * 16 + j + ((lane >> 4) << 3);
      int atom = row / 3;
      if (atom0 + atom < nAtoms) {
        long o = vBase + (long)row * H + colb;
        v_out[o] = v[o] + avvS[atom * H + colb] * accU[i][j];
      }
    }
  }
}

// Pre-pass: dst[n*K + k] = (bf16) src[k*N + n]   (read-coalesced transpose)
__global__ __launch_bounds__(256)
void convert_transpose_kernel(bf16* __restrict__ dst, const float* __restrict__ src,
                              int K, int N) {
  long total = (long)K * N;
  for (long idx = (long)blockIdx.x * blockDim.x + threadIdx.x; idx < total;
       idx += (long)gridDim.x * blockDim.x) {
    long k = idx / N, n = idx % N;
    dst[n * (long)K + k] = (bf16)src[idx];
  }
}

extern "C" void kernel_launch(void* const* d_in, const int* in_sizes, int n_in,
                              void* d_out, int out_size, void* d_ws, size_t ws_size,
                              hipStream_t stream) {
  const float* s     = (const float*)d_in[0];
  const float* v     = (const float*)d_in[1];
  const float* gamma = (const float*)d_in[2];
  const float* beta  = (const float*)d_in[3];
  const float* U_W   = (const float*)d_in[4];
  const float* V_W   = (const float*)d_in[5];
  const float* W1    = (const float*)d_in[6];
  const float* b1    = (const float*)d_in[7];
  const float* W2    = (const float*)d_in[8];
  const float* b2    = (const float*)d_in[9];

  int nAtoms = in_sizes[0] / H;
  float* s_out = (float*)d_out;
  float* v_out = s_out + (long)nAtoms * H;
  int blocks = (nAtoms + MT - 1) / MT;

  if (d_ws != nullptr && ws_size >= (size_t)WS_NEED_BYTES) {
    bf16* wsU  = (bf16*)d_ws + WSE_U;
    bf16* wsV  = (bf16*)d_ws + WSE_V;
    bf16* wsW1 = (bf16*)d_ws + WSE_W1;
    bf16* wsW2 = (bf16*)d_ws + WSE_W2;
    convert_transpose_kernel<<<64, 256, 0, stream>>>(wsU,  U_W, H,     H);
    convert_transpose_kernel<<<64, 256, 0, stream>>>(wsV,  V_W, H,     H);
    convert_transpose_kernel<<<64, 256, 0, stream>>>(wsW1, W1,  2 * H, H);
    convert_transpose_kernel<<<64, 256, 0, stream>>>(wsW2, W2,  H, 3 * H);
    painn_fused_kernel<true><<<blocks, 256, 0, stream>>>(
        s, v, gamma, beta, U_W, V_W, W1, b1, W2, b2,
        wsU, wsV, wsW1, wsW2, s_out, v_out, nAtoms);
  } else {
    painn_fused_kernel<false><<<blocks, 256, 0, stream>>>(
        s, v, gamma, beta, U_W, V_W, W1, b1, W2, b2,
        nullptr, nullptr, nullptr, nullptr, s_out, v_out, nAtoms);
  }
}
